// GNN_31903017075239
// MI455X (gfx1250) — compile-verified, matrix operand order checked by reading, not run on hardware
//
#include <hip/hip_runtime.h>
#include <hip/hip_bf16.h>
#include <math.h>

// ---------------------------------------------------------------------------
// GNN (GraphNorm + GraphConv x3 + mean-pool + MLP head) for MI455X (gfx1250).
// GEMMs run on v_wmma_f32_16x16x32_bf16 with split-bf16 (hi+lo) inputs for
// near-fp32 accuracy. Edge aggregation uses f32 global atomics. GraphNorm and
// pooling are deterministic segmented reductions over the sorted batch vector.
// ---------------------------------------------------------------------------

#define N_NODES 50000
#define N_EDGES 600000
#define N_GRAPHS 512
#define F_IN 64
#define HDIM 128
#define EPS 1e-5f

typedef __bf16 v16bf __attribute__((ext_vector_type(16)));
typedef float  v8f   __attribute__((ext_vector_type(8)));

// ---------------------------------------------------------------------------
// Graph segment boundaries: batch is sorted; gstart[g] = lower_bound(batch, g)
// ---------------------------------------------------------------------------
__global__ void k_gstart(const int* __restrict__ batch, int* __restrict__ gstart) {
    int g = blockIdx.x * blockDim.x + threadIdx.x;
    if (g > N_GRAPHS) return;
    if (g == N_GRAPHS) { gstart[N_GRAPHS] = N_NODES; return; }
    int lo = 0, hi = N_NODES;
    while (lo < hi) {
        int mid = (lo + hi) >> 1;
        if (batch[mid] < g) lo = mid + 1; else hi = mid;
    }
    gstart[g] = lo;
}

// ---------------------------------------------------------------------------
// PyG GraphNorm: out = w*(h - ms*mean)/sqrt(mean((h-ms*mean)^2)+eps) + b
// One block per graph, one thread per feature (deterministic, exact).
// ---------------------------------------------------------------------------
__global__ void k_gnorm(const float* __restrict__ src, float* __restrict__ dst,
                        const int* __restrict__ gstart,
                        const float* __restrict__ w, const float* __restrict__ b,
                        const float* __restrict__ ms, int F) {
    int g = blockIdx.x;
    int f = threadIdx.x;
    int n0 = gstart[g], n1 = gstart[g + 1];
    float cnt = fmaxf((float)(n1 - n0), 1.0f);
    float s = 0.0f;
    for (int n = n0; n < n1; ++n) s += src[(long)n * F + f];
    float shift = ms[f] * (s / cnt);          // ms * mean
    float v = 0.0f;
    for (int n = n0; n < n1; ++n) {
        float t = src[(long)n * F + f] - shift;
        v += t * t;
    }
    v /= cnt;
    float scale = w[f] / sqrtf(v + EPS);
    float bias = b[f];
    for (int n = n0; n < n1; ++n)
        dst[(long)n * F + f] = (src[(long)n * F + f] - shift) * scale + bias;
}

// ---------------------------------------------------------------------------
// Zero a float buffer
// ---------------------------------------------------------------------------
__global__ void k_zero(float* __restrict__ p, long n) {
    long i = (long)blockIdx.x * blockDim.x + threadIdx.x;
    if (i < n) p[i] = 0.0f;
}

// ---------------------------------------------------------------------------
// Edge scatter-add: agg[dst] += h[src]   (feature-parallel, coalesced)
// ---------------------------------------------------------------------------
__global__ void k_scatter(const float* __restrict__ h, float* __restrict__ agg,
                          const int* __restrict__ esrc, const int* __restrict__ edst,
                          int Fsh) {
    long i = (long)blockIdx.x * blockDim.x + threadIdx.x;
    int F = 1 << Fsh;
    long total = (long)N_EDGES << Fsh;
    if (i >= total) return;
    int f = (int)(i & (F - 1));
    int e = (int)(i >> Fsh);
    int s = esrc[e], d = edst[e];
    float val = h[(long)s * F + f];
    __hip_atomic_fetch_add(&agg[(long)d * F + f], val,
                           __ATOMIC_RELAXED, __HIP_MEMORY_SCOPE_AGENT);
}

// ---------------------------------------------------------------------------
// Build transposed concatenated weights: WT[n][k] = (k<Ka ? wrel[k][n] : wroot[k-Ka][n])
// WT is [HDIM x (Ka+Kb)] row-major (row = output feature n).
// ---------------------------------------------------------------------------
__global__ void k_wt(const float* __restrict__ wrel, const float* __restrict__ wroot,
                     float* __restrict__ WT, int Ka, int Kb) {
    int Ktot = Ka + Kb;
    int idx = blockIdx.x * blockDim.x + threadIdx.x;
    if (idx >= HDIM * Ktot) return;
    int n = idx / Ktot;
    int k = idx - n * Ktot;
    float v = (k < Ka) ? wrel[(long)k * HDIM + n] : wroot[(long)(k - Ka) * HDIM + n];
    WT[idx] = v;
}

// ---------------------------------------------------------------------------
// WMMA GEMM:  Out[M x 128] = relu?( [Xa | Xb] @ W + bias )
// Xa: [M x Ka] f32, Xb: [M x Kb] f32 (K-concatenated; Ka,Kb multiples of 32 or 0)
// WT: [128 x (Ka+Kb)] f32 (transposed weights). Split-bf16: 3 WMMAs per K tile.
// Grid: (M/16, 2), block 128 threads = 4 waves; each wave owns a 16x16 tile.
// ---------------------------------------------------------------------------
struct Frag { v16bf hi, lo; };

__device__ inline Frag load_frag(const float* __restrict__ p) {
    // p points at K=kb of this lane's row; needs K kb..kb+7 and kb+16..kb+23
    float4 f0 = *(const float4*)(p);
    float4 f1 = *(const float4*)(p + 4);
    float4 f2 = *(const float4*)(p + 16);
    float4 f3 = *(const float4*)(p + 20);
    float a[16] = { f0.x, f0.y, f0.z, f0.w, f1.x, f1.y, f1.z, f1.w,
                    f2.x, f2.y, f2.z, f2.w, f3.x, f3.y, f3.z, f3.w };
    Frag r;
#pragma unroll
    for (int e = 0; e < 16; ++e) {
        __bf16 h = (__bf16)a[e];
        r.hi[e] = h;
        r.lo[e] = (__bf16)(a[e] - (float)h);
    }
    return r;
}

__global__ void __launch_bounds__(128)
k_gemm(const float* __restrict__ Xa, int Ka,
       const float* __restrict__ Xb, int Kb,
       const float* __restrict__ WT, const float* __restrict__ bias,
       float* __restrict__ Out, int doRelu) {
    const int lane = threadIdx.x & 31;
    const int wave = threadIdx.x >> 5;
    const int mt = blockIdx.x;
    const int nt = blockIdx.y * 4 + wave;
    const int l15 = lane & 15;
    const int hi8 = (lane >> 4) * 8;
    const int Ktot = Ka + Kb;

    v8f acc = {};
    const int arow = mt * 16 + l15;  // A-matrix row handled by this lane
    const int wrow = nt * 16 + l15;  // output column / WT row for this lane

    for (int koff = 0; koff < Ktot; koff += 32) {
        const float* ap;
        int astr, ac0;
        if (koff < Ka) { ap = Xa; astr = Ka; ac0 = koff; }
        else           { ap = Xb; astr = Kb; ac0 = koff - Ka; }
        Frag A = load_frag(ap + (long)arow * astr + ac0 + hi8);
        Frag B = load_frag(WT + (long)wrow * Ktot + koff + hi8);
        acc = __builtin_amdgcn_wmma_f32_16x16x32_bf16(false, A.hi, false, B.hi,
                                                      (short)0, acc, false, false);
        acc = __builtin_amdgcn_wmma_f32_16x16x32_bf16(false, A.hi, false, B.lo,
                                                      (short)0, acc, false, false);
        acc = __builtin_amdgcn_wmma_f32_16x16x32_bf16(false, A.lo, false, B.hi,
                                                      (short)0, acc, false, false);
    }

    float bc = bias ? bias[wrow] : 0.0f;
#pragma unroll
    for (int r = 0; r < 8; ++r) {
        float v = acc[r] + bc;
        if (doRelu) v = fmaxf(v, 0.0f);
        Out[(long)(mt * 16 + r + hi8) * HDIM + wrow] = v;
    }
}

// ---------------------------------------------------------------------------
// global_mean_pool: pooled[g][f] = mean over graph-g nodes of h[:,f]
// ---------------------------------------------------------------------------
__global__ void k_pool(const float* __restrict__ h, float* __restrict__ pooled,
                       const int* __restrict__ gstart) {
    int g = blockIdx.x;
    int f = threadIdx.x;
    int n0 = gstart[g], n1 = gstart[g + 1];
    float cnt = fmaxf((float)(n1 - n0), 1.0f);
    float s = 0.0f;
    for (int n = n0; n < n1; ++n) s += h[(long)n * HDIM + f];
    pooled[(long)g * HDIM + f] = s / cnt;
}

// ---------------------------------------------------------------------------
// Output head: logits = denseOut @ out_w + out_b ; softmax over C=2
// ---------------------------------------------------------------------------
__global__ void k_head(const float* __restrict__ dOut, const float* __restrict__ out_w,
                       const float* __restrict__ out_b, float* __restrict__ y) {
    int g = blockIdx.x * blockDim.x + threadIdx.x;
    if (g >= N_GRAPHS) return;
    float l0 = out_b[0], l1 = out_b[1];
    for (int k = 0; k < HDIM; ++k) {
        float hv = dOut[(long)g * HDIM + k];
        l0 += hv * out_w[k * 2 + 0];
        l1 += hv * out_w[k * 2 + 1];
    }
    float m = fmaxf(l0, l1);
    float e0 = __expf(l0 - m), e1 = __expf(l1 - m);
    float inv = 1.0f / (e0 + e1);
    y[(long)g * 2 + 0] = e0 * inv;
    y[(long)g * 2 + 1] = e1 * inv;
}

// ---------------------------------------------------------------------------
// Host-side orchestration
// ---------------------------------------------------------------------------
extern "C" void kernel_launch(void* const* d_in, const int* in_sizes, int n_in,
                              void* d_out, int out_size, void* d_ws, size_t ws_size,
                              hipStream_t stream) {
    // inputs
    const float* x        = (const float*)d_in[0];
    const int*   eidx     = (const int*)d_in[1];
    const int*   batch    = (const int*)d_in[2];
    const float* gn0_w    = (const float*)d_in[3];
    const float* gn0_b    = (const float*)d_in[4];
    const float* gn0_ms   = (const float*)d_in[5];
    const float* gn1_w    = (const float*)d_in[6];
    const float* gn1_b    = (const float*)d_in[7];
    const float* gn1_ms   = (const float*)d_in[8];
    const float* gn2_w    = (const float*)d_in[9];
    const float* gn2_b    = (const float*)d_in[10];
    const float* gn2_ms   = (const float*)d_in[11];
    const float* c1_wrel  = (const float*)d_in[12];
    const float* c1_brel  = (const float*)d_in[13];
    const float* c1_wroot = (const float*)d_in[14];
    const float* c2_wrel  = (const float*)d_in[15];
    const float* c2_brel  = (const float*)d_in[16];
    const float* c2_wroot = (const float*)d_in[17];
    const float* c3_wrel  = (const float*)d_in[18];
    const float* c3_brel  = (const float*)d_in[19];
    const float* c3_wroot = (const float*)d_in[20];
    const float* dense_w  = (const float*)d_in[21];
    const float* dense_b  = (const float*)d_in[22];
    const float* out_w    = (const float*)d_in[23];
    const float* out_b    = (const float*)d_in[24];
    const int* esrc = eidx;
    const int* edst = eidx + N_EDGES;

    // workspace carve-up (~78 MB)
    float* bufH   = (float*)d_ws;                       // [N_NODES x 128] activations
    float* bufH2  = bufH  + (long)N_NODES * HDIM;       // [N_NODES x 128] activations
    float* agg    = bufH2 + (long)N_NODES * HDIM;       // [N_NODES x 128] edge aggregate
    float* WT     = agg   + (long)N_NODES * HDIM;       // [128 x 256] transposed weights
    float* pooled = WT + HDIM * 256;                    // [512 x 128]
    float* dOut   = pooled + (long)N_GRAPHS * HDIM;     // [512 x 128]
    int*   gstart = (int*)(dOut + (long)N_GRAPHS * HDIM); // [513]

    // graph boundaries
    k_gstart<<<(N_GRAPHS + 1 + 255) / 256, 256, 0, stream>>>(batch, gstart);

    // gnorm0 on raw input (F=64) -> bufH [N x 64]
    k_gnorm<<<N_GRAPHS, F_IN, 0, stream>>>(x, bufH, gstart, gn0_w, gn0_b, gn0_ms, F_IN);

    const long nf64  = (long)N_NODES * 64;
    const long nf128 = (long)N_NODES * 128;

    // ---- conv1: [agg64 | h64] @ [c1_wrel; c1_wroot] + brel, relu -> bufH2 [N x 128]
    k_zero<<<(int)((nf64 + 255) / 256), 256, 0, stream>>>(agg, nf64);
    k_scatter<<<(int)((((long)N_EDGES << 6) + 255) / 256), 256, 0, stream>>>(
        bufH, agg, esrc, edst, 6);
    k_wt<<<(HDIM * 128 + 255) / 256, 256, 0, stream>>>(c1_wrel, c1_wroot, WT, 64, 64);
    k_gemm<<<dim3(N_NODES / 16, 2), 128, 0, stream>>>(agg, 64, bufH, 64, WT, c1_brel,
                                                      bufH2, 1);

    // gnorm1 in place on bufH2 (F=128)
    k_gnorm<<<N_GRAPHS, HDIM, 0, stream>>>(bufH2, bufH2, gstart, gn1_w, gn1_b, gn1_ms, HDIM);

    // ---- conv2 -> bufH
    k_zero<<<(int)((nf128 + 255) / 256), 256, 0, stream>>>(agg, nf128);
    k_scatter<<<(int)((((long)N_EDGES << 7) + 255) / 256), 256, 0, stream>>>(
        bufH2, agg, esrc, edst, 7);
    k_wt<<<(HDIM * 256 + 255) / 256, 256, 0, stream>>>(c2_wrel, c2_wroot, WT, 128, 128);
    k_gemm<<<dim3(N_NODES / 16, 2), 128, 0, stream>>>(agg, 128, bufH2, 128, WT, c2_brel,
                                                      bufH, 1);

    // gnorm2 in place on bufH (F=128)
    k_gnorm<<<N_GRAPHS, HDIM, 0, stream>>>(bufH, bufH, gstart, gn2_w, gn2_b, gn2_ms, HDIM);

    // ---- conv3 -> bufH2
    k_zero<<<(int)((nf128 + 255) / 256), 256, 0, stream>>>(agg, nf128);
    k_scatter<<<(int)((((long)N_EDGES << 7) + 255) / 256), 256, 0, stream>>>(
        bufH, agg, esrc, edst, 7);
    k_wt<<<(HDIM * 256 + 255) / 256, 256, 0, stream>>>(c3_wrel, c3_wroot, WT, 128, 128);
    k_gemm<<<dim3(N_NODES / 16, 2), 128, 0, stream>>>(agg, 128, bufH, 128, WT, c3_brel,
                                                      bufH2, 1);

    // ---- mean pool -> pooled [512 x 128]
    k_pool<<<N_GRAPHS, HDIM, 0, stream>>>(bufH2, pooled, gstart);

    // ---- dense: relu(pooled @ dense_w + dense_b) -> dOut [512 x 128]
    k_wt<<<(HDIM * 128 + 255) / 256, 256, 0, stream>>>(dense_w, nullptr, WT, 128, 0);
    k_gemm<<<dim3(N_GRAPHS / 16, 2), 128, 0, stream>>>(pooled, 128, nullptr, 0, WT,
                                                       dense_b, dOut, 1);

    // ---- out layer + softmax -> d_out [512 x 2]
    k_head<<<(N_GRAPHS + 255) / 256, 256, 0, stream>>>(dOut, out_w, out_b, (float*)d_out);
}
